// RandomSetLoss_9079560864659
// MI455X (gfx1250) — matrix operand order repaired and checked by reading, not compiled, not run on hardware
//
#include <hip/hip_runtime.h>
#include <hip/hip_bf16.h>

#define BATCH       16384
#define NUM_SETS    1024
#define NUM_CLASSES 128
#define ALPHA       0.001f
#define BETA        0.001f

typedef __attribute__((ext_vector_type(16))) __bf16          v16bf;
typedef __attribute__((ext_vector_type(8)))  float           v8f;
typedef __attribute__((ext_vector_type(8)))  unsigned short  us8v;

union FragBF {
    v16bf v;
    us8v  h[2];
    unsigned short us[16];
};

// Native f32 -> bf16 (RNE) — backend selects hardware cvt ops.
__device__ __forceinline__ unsigned short f2bf(float f) {
    union { __bf16 h; unsigned short u; } cv;
    cv.h = (__bf16)f;
    return cv.u;
}

// ---------------------------------------------------------------------------
// Workspace layout:
//   floats:  ws[0]=bce_sum, ws[1]=mr_sum, ws[16..16+BATCH) = rowsum
//   bytes:   [kPbOff, +32MB) = bf16 copy of clipped pred
//            [kMbOff, +2MB)  = bf16 copy of moebius
// ---------------------------------------------------------------------------
static constexpr size_t kPbOff   = 66048;                            // 256-aligned
static constexpr size_t kPbBytes = (size_t)BATCH * NUM_SETS * 2;     // 32 MB
static constexpr size_t kMbOff   = kPbOff + kPbBytes;
static constexpr size_t kMbBytes = (size_t)NUM_SETS * NUM_SETS * 2;  // 2 MB
static constexpr size_t kWsNeed  = kMbOff + kMbBytes;

__global__ void zero_ws_kernel(float* ws, int n) {
    int i = blockIdx.x * blockDim.x + threadIdx.x;
    if (i < n) ws[i] = 0.0f;
}

// ---------------------------------------------------------------------------
// BCE + fused f32->bf16 conversion of clipped p (8 elements / thread).
// ---------------------------------------------------------------------------
__global__ __launch_bounds__(256)
void bce_convert_kernel(const float* __restrict__ pred,
                        const float* __restrict__ membership,
                        const int*   __restrict__ tgt,
                        float*       __restrict__ ws,
                        unsigned short* __restrict__ pbf) {  // may be null
    const int nvec = (BATCH * NUM_SETS) / 8;   // 2,097,152
    const int step = gridDim.x * blockDim.x;
    float acc = 0.0f;
    for (int v = blockIdx.x * blockDim.x + threadIdx.x; v < nvec; v += step) {
        int b  = v >> 7;               // (v*8) / NUM_SETS
        int s0 = (v & 127) * 8;
        int cls = tgt[b] & (NUM_CLASSES - 1);
        const float4* pr = (const float4*)(pred       + (size_t)b   * NUM_SETS + s0);
        const float4* mr = (const float4*)(membership + (size_t)cls * NUM_SETS + s0);
        float4 p0 = pr[0], p1 = pr[1];
        float4 t0 = mr[0], t1 = mr[1];
        float pv[8] = {p0.x, p0.y, p0.z, p0.w, p1.x, p1.y, p1.z, p1.w};
        float tv[8] = {t0.x, t0.y, t0.z, t0.w, t1.x, t1.y, t1.z, t1.w};
        us8v r;
        #pragma unroll
        for (int j = 0; j < 8; ++j) {
            float p = fminf(fmaxf(pv[j], 1e-7f), 1.0f - 1e-7f);
            acc -= tv[j] * __logf(p) + (1.0f - tv[j]) * __logf(1.0f - p);
            r[j] = f2bf(p);
        }
        if (pbf) ((us8v*)(pbf))[v] = r;
    }
    #pragma unroll
    for (int off = 16; off; off >>= 1) acc += __shfl_xor(acc, off, 32);
    if ((threadIdx.x & 31) == 0) atomicAdd(&ws[0], acc);
}

__global__ __launch_bounds__(256)
void convert_m_kernel(const float* __restrict__ Mo, unsigned short* __restrict__ mbf) {
    int v = blockIdx.x * blockDim.x + threadIdx.x;
    const int nvec = (NUM_SETS * NUM_SETS) / 8;
    if (v >= nvec) return;
    const float4* q = (const float4*)(Mo + (size_t)v * 8);
    float4 x0 = q[0], x1 = q[1];
    float xv[8] = {x0.x, x0.y, x0.z, x0.w, x1.x, x1.y, x1.z, x1.w};
    us8v r;
    #pragma unroll
    for (int j = 0; j < 8; ++j) r[j] = f2bf(xv[j]);
    ((us8v*)mbf)[v] = r;
}

// ---------------------------------------------------------------------------
// Shared epilogue over 2(M) x 4(N) accumulator tiles.
// C/D layout: elem e of lane L is row m0+i*16+e+8*(L>>4), col n0+j*16+(L&15).
// ---------------------------------------------------------------------------
__device__ __forceinline__ void loss_epilogue(v8f acc[2][4], float* rs,
                                              float* ws, int lane, int l15,
                                              int lhi, int wm) {
    float mrp = 0.0f;
    #pragma unroll
    for (int i = 0; i < 2; ++i)
        #pragma unroll
        for (int j = 0; j < 4; ++j)
            #pragma unroll
            for (int e = 0; e < 8; ++e)
                mrp += fmaxf(-acc[i][j][e], 0.0f);
    #pragma unroll
    for (int off = 16; off; off >>= 1) mrp += __shfl_xor(mrp, off, 32);
    if (lane == 0) atomicAdd(&ws[1], mrp);

    #pragma unroll
    for (int i = 0; i < 2; ++i) {
        #pragma unroll
        for (int e = 0; e < 8; ++e) {
            float v = (acc[i][0][e] + acc[i][1][e]) + (acc[i][2][e] + acc[i][3][e]);
            #pragma unroll
            for (int off = 8; off; off >>= 1) v += __shfl_xor(v, off, 16);
            if (l15 == 0) atomicAdd(&rs[wm * 32 + i * 16 + e + 8 * lhi], v);
        }
    }
}

// ---------------------------------------------------------------------------
// Fast GEMM: bf16 fragment loads, no conversion VALU in the K loop.
// Block 64(M) x 256(N), 8 waves (2x4), wave tile 32(M) x 64(N) = 2x4 WMMA tiles.
// Per K-step: 12 x global_load_b128 -> 8 x v_wmma (1.5 loads/WMMA).
// ---------------------------------------------------------------------------
__global__ __launch_bounds__(256)
void gemm_loss_bf16_kernel(const unsigned short* __restrict__ Pb,
                           const unsigned short* __restrict__ Mb,
                           float* __restrict__ ws) {
    __shared__ float rs[64];
    const int tid = threadIdx.x;
    if (tid < 64) rs[tid] = 0.0f;
    __syncthreads();

    const int lane = tid & 31;
    const int w    = tid >> 5;
    const int wm   = w >> 2;     // 0..1
    const int wn   = w & 3;      // 0..3
    const int blockM = blockIdx.y * 64;
    const int blockN = blockIdx.x * 256;
    const int m0 = blockM + wm * 32;
    const int n0 = blockN + wn * 64;

    const int l15 = lane & 15;
    const int lhi = lane >> 4;
    const int c0  = lhi * 8;     // A K sub-offset (elements)
    const int kb  = lhi * 8;     // B K sub-offset (elements)

    // A: lane holds row m0+l15 (+16 for 2nd tile), K = c0..c0+7 and c0+16..c0+23
    const unsigned short* ap[2];
    ap[0] = Pb + (size_t)(m0      + l15) * NUM_SETS + c0;
    ap[1] = Pb + (size_t)(m0 + 16 + l15) * NUM_SETS + c0;
    // B: lane holds column n0+j*16+l15 (= row of M), K = kb..kb+7 and kb+16..kb+23
    const unsigned short* bp[4];
    #pragma unroll
    for (int j = 0; j < 4; ++j)
        bp[j] = Mb + (size_t)(n0 + j * 16 + l15) * NUM_SETS + kb;

    const v8f zero8 = {0.f, 0.f, 0.f, 0.f, 0.f, 0.f, 0.f, 0.f};
    v8f acc[2][4];
    #pragma unroll
    for (int i = 0; i < 2; ++i)
        #pragma unroll
        for (int j = 0; j < 4; ++j) acc[i][j] = zero8;

    for (int k = 0; k < NUM_SETS; k += 32) {
        if (k + 32 < NUM_SETS) {   // WGP-scope prefetch of next K chunk
            __builtin_prefetch(ap[0] + k + 32, 0, 3);
            __builtin_prefetch(bp[0] + k + 32, 0, 3);
        }
        FragBF a[2], b[4];
        #pragma unroll
        for (int i = 0; i < 2; ++i) {
            a[i].h[0] = ((const us8v*)(ap[i] + k))[0];
            a[i].h[1] = ((const us8v*)(ap[i] + k))[2];   // +16 elements
        }
        #pragma unroll
        for (int j = 0; j < 4; ++j) {
            b[j].h[0] = ((const us8v*)(bp[j] + k))[0];
            b[j].h[1] = ((const us8v*)(bp[j] + k))[2];   // +16 elements
        }
        #pragma unroll
        for (int i = 0; i < 2; ++i)
            #pragma unroll
            for (int j = 0; j < 4; ++j)
                acc[i][j] = __builtin_amdgcn_wmma_f32_16x16x32_bf16(
                    false, a[i].v, false, b[j].v, (short)0, acc[i][j], false, false);
    }

    loss_epilogue(acc, rs, ws, lane, l15, lhi, wm);
    __syncthreads();
    if (tid < 64) atomicAdd(&ws[16 + blockM + tid], rs[tid]);
}

// ---------------------------------------------------------------------------
// Fallback GEMM (workspace too small): convert in-loop via native __bf16 casts,
// clamping A (reference matmuls the clipped p).
// ---------------------------------------------------------------------------
__device__ __forceinline__ void pack4f(FragBF& f, int base, float4 x, bool clip) {
    if (clip) {
        x.x = fminf(fmaxf(x.x, 1e-7f), 1.0f - 1e-7f);
        x.y = fminf(fmaxf(x.y, 1e-7f), 1.0f - 1e-7f);
        x.z = fminf(fmaxf(x.z, 1e-7f), 1.0f - 1e-7f);
        x.w = fminf(fmaxf(x.w, 1e-7f), 1.0f - 1e-7f);
    }
    f.us[base + 0] = f2bf(x.x);
    f.us[base + 1] = f2bf(x.y);
    f.us[base + 2] = f2bf(x.z);
    f.us[base + 3] = f2bf(x.w);
}

__global__ __launch_bounds__(256)
void gemm_loss_f32_kernel(const float* __restrict__ P,
                          const float* __restrict__ Mo,
                          float* __restrict__ ws) {
    __shared__ float rs[64];
    const int tid = threadIdx.x;
    if (tid < 64) rs[tid] = 0.0f;
    __syncthreads();

    const int lane = tid & 31;
    const int w    = tid >> 5;
    const int wm   = w >> 2;
    const int wn   = w & 3;
    const int blockM = blockIdx.y * 64;
    const int blockN = blockIdx.x * 256;
    const int m0 = blockM + wm * 32;
    const int n0 = blockN + wn * 64;

    const int l15 = lane & 15;
    const int lhi = lane >> 4;
    const int c0  = lhi * 8;
    const int kb  = lhi * 8;

    const float* ap[2];
    ap[0] = P + (size_t)(m0      + l15) * NUM_SETS + c0;
    ap[1] = P + (size_t)(m0 + 16 + l15) * NUM_SETS + c0;
    const float* bp[4];
    #pragma unroll
    for (int j = 0; j < 4; ++j)
        bp[j] = Mo + (size_t)(n0 + j * 16 + l15) * NUM_SETS + kb;

    const v8f zero8 = {0.f, 0.f, 0.f, 0.f, 0.f, 0.f, 0.f, 0.f};
    v8f acc[2][4];
    #pragma unroll
    for (int i = 0; i < 2; ++i)
        #pragma unroll
        for (int j = 0; j < 4; ++j) acc[i][j] = zero8;

    for (int k = 0; k < NUM_SETS; k += 32) {
        FragBF a[2], b[4];
        #pragma unroll
        for (int i = 0; i < 2; ++i) {
            const float4* q = (const float4*)(ap[i] + k);
            pack4f(a[i], 0,  q[0], true);
            pack4f(a[i], 4,  q[1], true);
            pack4f(a[i], 8,  q[4], true);   // K +16..19
            pack4f(a[i], 12, q[5], true);   // K +20..23
        }
        #pragma unroll
        for (int j = 0; j < 4; ++j) {
            const float4* q = (const float4*)(bp[j] + k);
            pack4f(b[j], 0,  q[0], false);
            pack4f(b[j], 4,  q[1], false);
            pack4f(b[j], 8,  q[4], false);
            pack4f(b[j], 12, q[5], false);
        }
        #pragma unroll
        for (int i = 0; i < 2; ++i)
            #pragma unroll
            for (int j = 0; j < 4; ++j)
                acc[i][j] = __builtin_amdgcn_wmma_f32_16x16x32_bf16(
                    false, a[i].v, false, b[j].v, (short)0, acc[i][j], false, false);
    }

    loss_epilogue(acc, rs, ws, lane, l15, lhi, wm);
    __syncthreads();
    if (tid < 64) atomicAdd(&ws[16 + blockM + tid], rs[tid]);
}

__global__ __launch_bounds__(256)
void finalize_kernel(const float* __restrict__ ws, float* __restrict__ out) {
    __shared__ float red[8];
    float acc = 0.0f;
    for (int i = threadIdx.x; i < BATCH; i += blockDim.x)
        acc += fabsf(ws[16 + i] - 1.0f);
    #pragma unroll
    for (int off = 16; off; off >>= 1) acc += __shfl_xor(acc, off, 32);
    if ((threadIdx.x & 31) == 0) red[threadIdx.x >> 5] = acc;
    __syncthreads();
    if (threadIdx.x == 0) {
        float ms_sum = 0.0f;
        #pragma unroll
        for (int i = 0; i < 8; ++i) ms_sum += red[i];
        const float invBS = 1.0f / ((float)BATCH * (float)NUM_SETS);
        float bce = ws[0] * invBS;
        float mr  = ws[1] * invBS;
        float ms  = ms_sum / (float)BATCH;
        out[0] = bce + ALPHA * mr + BETA * ms;
        out[1] = bce;
        out[2] = mr;
        out[3] = ms;
    }
}

extern "C" void kernel_launch(void* const* d_in, const int* in_sizes, int n_in,
                              void* d_out, int out_size, void* d_ws, size_t ws_size,
                              hipStream_t stream) {
    const float* pred = (const float*)d_in[0];
    const float* memb = (const float*)d_in[1];
    const float* moeb = (const float*)d_in[2];
    const int*   tgt  = (const int*)d_in[3];
    float* out = (float*)d_out;
    float* ws  = (float*)d_ws;

    (void)in_sizes; (void)n_in; (void)out_size;

    const bool fast = (ws_size >= kWsNeed);
    unsigned short* pbf = fast ? (unsigned short*)((char*)d_ws + kPbOff) : nullptr;
    unsigned short* mbf = fast ? (unsigned short*)((char*)d_ws + kMbOff) : nullptr;

    const int nz = 16 + BATCH;
    zero_ws_kernel<<<(nz + 255) / 256, 256, 0, stream>>>(ws, nz);

    bce_convert_kernel<<<2048, 256, 0, stream>>>(pred, memb, tgt, ws, pbf);

    dim3 grid(NUM_SETS / 256, BATCH / 64);   // (4, 256)
    if (fast) {
        convert_m_kernel<<<(NUM_SETS * NUM_SETS / 8 + 255) / 256, 256, 0, stream>>>(moeb, mbf);
        gemm_loss_bf16_kernel<<<grid, 256, 0, stream>>>(pbf, mbf, ws);
    } else {
        gemm_loss_f32_kernel<<<grid, 256, 0, stream>>>(pred, moeb, ws);
    }

    finalize_kernel<<<1, 256, 0, stream>>>(ws, out);
}
